// GraphDecodingNet_32787780338230
// MI455X (gfx1250) — matrix operand order repaired and checked by reading, not compiled, not run on hardware
//
#include <hip/hip_runtime.h>

#define LAG   5
#define HID   256
#define NA    30
#define NE    870
#define BB    2
#define TT    200
#define TMM   (TT - LAG)          // 195
#define EPB   (NA - 1)            // 29 edges per receiver (contiguous blocks)
#define CHUNK 32
#define NCHUNK ((NE + CHUNK - 1) / CHUNK)   // 28
#define EPAD  (NCHUNK * CHUNK)              // 896

typedef __attribute__((ext_vector_type(16))) _Float16 v16h;
typedef __attribute__((ext_vector_type(8)))  float    v8f;

#define WMMA_F16(A, B, C) \
  __builtin_amdgcn_wmma_f32_16x16x32_f16(false, (A), false, (B), (short)0, (C), false, false)

// ---- WMMA fragment coordinate maps (CDNA5 ISA 7.12.2, wave32) -------------
// 16-bit A matrix 16x32: lanes 0-15 -> M=lane, K in {0..7,16..23};
//                        lanes 16-31 -> M=lane-16, K in {8..15,24..31}
__device__ __forceinline__ void a_coord(int lane, int i, int& m, int& k) {
  int v = i >> 1, hb = i & 1;
  m = lane & 15;
  int kb = (v < 4) ? (2 * v + hb) : (16 + 2 * (v - 4) + hb);
  k = kb + ((lane & 16) ? 8 : 0);
}
// inverse: (m_loc 0..15, k_loc 0..31) -> half offset (lane*16 + i) within a frag
__device__ __forceinline__ int inv_a(int m_loc, int k_loc) {
  int lane_f = (m_loc & 15) | (((k_loc >> 3) & 1) << 4);
  int kk = k_loc & 0x17;                       // drop bit3
  int i  = (kk & 7) | ((kk & 16) >> 1);
  return lane_f * 16 + i;
}
// 16-bit B matrix 32x16: lanes 0-15 -> N=lane, K in {0..7,16..23}; lanes 16-31 -> N=lane-16, K in {8..15,24..31}
__device__ __forceinline__ void b_coord(int lane, int i, int& k, int& n) {
  n = lane & 15;
  k = ((i < 8) ? i : (i + 8)) + ((lane & 16) ? 8 : 0);
}
// inverse: (k_loc 0..31, n_loc 0..15) -> half offset within a frag
__device__ __forceinline__ int inv_b(int k_loc, int n_loc) {
  return ((n_loc & 15) | (((k_loc >> 3) & 1) << 4)) * 16 + ((k_loc & 7) | ((k_loc & 16) >> 1));
}

// One wave: 32x32 tile of a 32x256 @ 256x256 GEMM (A frags in LDS, B frags packed global)
__device__ __forceinline__ void wave_gemm(const _Float16* sA, const _Float16* wp,
                                          int nt0, int lane, v8f acc[2][2]) {
#pragma unroll
  for (int kt = 0; kt < 8; ++kt) {
    v16h a0 = *(const v16h*)(sA + (0 * 8 + kt) * 512 + lane * 16);
    v16h a1 = *(const v16h*)(sA + (1 * 8 + kt) * 512 + lane * 16);
    v16h b0 = *(const v16h*)(wp + (kt * 16 + nt0) * 512 + lane * 16);
    v16h b1 = *(const v16h*)(wp + (kt * 16 + nt0 + 1) * 512 + lane * 16);
    acc[0][0] = WMMA_F16(a0, b0, acc[0][0]);
    acc[0][1] = WMMA_F16(a0, b1, acc[0][1]);
    acc[1][0] = WMMA_F16(a1, b0, acc[1][0]);
    acc[1][1] = WMMA_F16(a1, b1, acc[1][1]);
  }
}

// ---- kernel 1: edge reparameterization ------------------------------------
__global__ void edge_scale_kernel(const float* __restrict__ mu,
                                  const float* __restrict__ logvar,
                                  const float* __restrict__ eps,
                                  float* __restrict__ esc) {
  int i = blockIdx.x * 256 + threadIdx.x;
  if (i < BB * NE) esc[i] = eps[i] * __expf(0.5f * logvar[i]) + mu[i];
}

// ---- kernel 2: pack weights into B-fragment layout ------------------------
// blocks 0..127: W2 | 128..255: W_out1 rows 5.. | 256..383: W_out2 | 384..399: W1 (K padded 10->32)
__global__ __launch_bounds__(32)
void pack_weights_kernel(const float* __restrict__ w_msg1,
                         const float* __restrict__ w_msg2,
                         const float* __restrict__ w_out1,
                         const float* __restrict__ w_out2,
                         _Float16* __restrict__ pW1,
                         _Float16* __restrict__ pW2,
                         _Float16* __restrict__ pO1,
                         _Float16* __restrict__ pO2) {
  int bx = blockIdx.x;
  int lane = threadIdx.x;
  if (bx >= 384) {                       // W1: single K-tile, 16 N-tiles
    int nt = bx - 384;
#pragma unroll
    for (int i = 0; i < 16; ++i) {
      int k, n; b_coord(lane, i, k, n);
      float v = (k < 2 * LAG) ? w_msg1[k * HID + nt * 16 + n] : 0.f;
      pW1[nt * 512 + lane * 16 + i] = (_Float16)v;
    }
    return;
  }
  int mat  = bx >> 7;
  int frag = bx & 127;
  int kt = frag >> 4, nt = frag & 15;
  const float* src; _Float16* dst; int rowoff = 0;
  if (mat == 0)      { src = w_msg2; dst = pW2; }
  else if (mat == 1) { src = w_out1; dst = pO1; rowoff = LAG; }
  else               { src = w_out2; dst = pO2; }
#pragma unroll
  for (int i = 0; i < 16; ++i) {
    int k, n; b_coord(lane, i, k, n);
    dst[frag * 512 + lane * 16 + i] =
        (_Float16)src[(kt * 32 + k + rowoff) * HID + (nt * 16 + n)];
  }
}

// ---- kernel 3: fused NRI decoder, one workgroup per (b,t) -----------------
#define SMEM_BYTES 51840

__global__ __launch_bounds__(256)
void nri_decoder_kernel(const float* __restrict__ inputs,
                        const float* __restrict__ escale,
                        const float* __restrict__ b_msg1,
                        const float* __restrict__ b_msg2,
                        const float* __restrict__ w_out1,
                        const float* __restrict__ b_out1,
                        const float* __restrict__ b_out2,
                        const float* __restrict__ w_out3,
                        const float* __restrict__ b_out3,
                        const _Float16* __restrict__ pW1,
                        const _Float16* __restrict__ pW2,
                        const _Float16* __restrict__ pO1,
                        const _Float16* __restrict__ pO2,
                        float* __restrict__ out) {
  extern __shared__ char smem[];
  float*    sData = (float*)smem;            // [32][8] lag window (padded)
  float*    sEsc  = sData + 32 * 8;          // [896] edge scales (zero padded)
  float*    sB1   = sEsc + EPAD;             // b_msg1
  float*    sB2   = sB1 + HID;               // b_msg2
  float*    sBo1  = sB2 + HID;               // b_out1
  float*    sBo2  = sBo1 + HID;              // b_out2
  float*    sWo1d = sBo2 + HID;              // w_out1 rows 0..4 (lag part of aug)
  float*    sWo3  = sWo1d + LAG * HID;       // w_out3
  float*    sRed  = sWo3 + HID;              // [32] out3 reduction
  _Float16* sPM   = (_Float16*)(sRed + 32);  // pre_msg A-frags: 2 x 512
  _Float16* sEA   = sPM + 1024;              // edge-scale agg A-frags: 2 x 512
  _Float16* sFA   = sEA + 1024;              // 16 A-frags (h1 / agg / ...)
  _Float16* sFB   = sFA + 8192;              // 16 B-frags (msg) / A-frags (pred1)

  const int tid  = threadIdx.x;
  const int lane = tid & 31;
  const int wave = tid >> 5;
  const int t = blockIdx.x;
  const int b = blockIdx.y;
  const int nt0 = wave * 2;

  // prefetch packed weights toward L2 (global_prefetch_b8)
  __builtin_prefetch(pW2 + (size_t)tid * 256, 0, 1);
  __builtin_prefetch(pO1 + (size_t)tid * 256, 0, 1);
  __builtin_prefetch(pO2 + (size_t)tid * 256, 0, 1);

  // ---- stage block-local state ----
  for (int i = tid; i < HID; i += 256) {
    sB1[i] = b_msg1[i]; sB2[i] = b_msg2[i];
    sBo1[i] = b_out1[i]; sBo2[i] = b_out2[i]; sWo3[i] = w_out3[i];
  }
  for (int i = tid; i < LAG * HID; i += 256) sWo1d[i] = w_out1[i];
  for (int i = tid; i < 32 * 8; i += 256) {
    int n = i >> 3, l = i & 7;
    sData[i] = (n < NA && l < LAG) ? inputs[(b * NA + n) * TT + (t + l)] : 0.f;
  }
  for (int i = tid; i < EPAD; i += 256) sEsc[i] = (i < NE) ? escale[b * NE + i] : 0.f;
  if (tid < 32) sRed[tid] = 0.f;
  __syncthreads();

  v8f z8 = {0.f, 0.f, 0.f, 0.f, 0.f, 0.f, 0.f, 0.f};
  v8f aggAcc[2][2] = {{z8, z8}, {z8, z8}};   // agg[node][h] lives in registers

  const int hi8 = (lane & 16) ? 8 : 0;
  const int ncol = lane & 15;

  // ---- edge-message pipeline: 28 chunks of 32 edges ----
  for (int c = 0; c < NCHUNK; ++c) {
    const int base = c * CHUNK;

    // Phase A: build pre_msg A-tile (32x32, K=10 real) and edge-scale agg A-tile
    for (int idx = tid; idx < 1024; idx += 256) {
      int mt = idx >> 9, fl = (idx >> 4) & 31, fi = idx & 15;
      int ml, kl; a_coord(fl, fi, ml, kl);
      int e = base + mt * 16 + ml;
      float v = 0.f;
      if (e < NE) {
        int rcv = e / EPB, js = e - rcv * EPB;
        int snd = js + (js >= rcv);
        v = (kl < LAG) ? sData[snd * 8 + kl]
                       : (kl < 2 * LAG ? sData[rcv * 8 + (kl - LAG)] : 0.f);
      }
      sPM[idx] = (_Float16)v;
    }
    for (int idx = tid; idx < 1024; idx += 256) {
      int mt = idx >> 9, fl = (idx >> 4) & 31, fi = idx & 15;
      int ml, kl; a_coord(fl, fi, ml, kl);
      int m = mt * 16 + ml;               // node
      int e = base + kl;                  // edge (padded edges have scale 0)
      sEA[idx] = (_Float16)(((e / EPB) == m) ? sEsc[e] : 0.f);
    }
    __syncthreads();

    // Phase B: msg1 on tensor pipe (K=32), epilogue -> h1 A-frags (f16)
    {
      v16h a0 = *(const v16h*)(sPM + 0 * 512 + lane * 16);
      v16h a1 = *(const v16h*)(sPM + 1 * 512 + lane * 16);
      v16h b0 = *(const v16h*)(pW1 + (nt0)     * 512 + lane * 16);
      v16h b1 = *(const v16h*)(pW1 + (nt0 + 1) * 512 + lane * 16);
      v8f h1[2][2];
      h1[0][0] = WMMA_F16(a0, b0, z8);
      h1[0][1] = WMMA_F16(a0, b1, z8);
      h1[1][0] = WMMA_F16(a1, b0, z8);
      h1[1][1] = WMMA_F16(a1, b1, z8);
#pragma unroll
      for (int mt = 0; mt < 2; ++mt)
#pragma unroll
        for (int nn = 0; nn < 2; ++nn) {
          int n = (nt0 + nn) * 16 + ncol;           // hid1 index = msg2 K
#pragma unroll
          for (int r = 0; r < 8; ++r) {
            int m_loc = r + hi8;
            float v = h1[mt][nn][r] + sB1[n];
            v = v > 0.f ? v : 0.f;
            sFA[(mt * 8 + (n >> 5)) * 512 + inv_a(m_loc, n & 31)] = (_Float16)v;
          }
        }
    }
    __syncthreads();

    // Phase C: msg2 GEMM (32 WMMA), epilogue relu+bias -> msg B-frags (f16)
    {
      v8f acc[2][2] = {{z8, z8}, {z8, z8}};
      wave_gemm(sFA, pW2, nt0, lane, acc);
#pragma unroll
      for (int mt = 0; mt < 2; ++mt)
#pragma unroll
        for (int nn = 0; nn < 2; ++nn) {
          int n = (nt0 + nn) * 16 + ncol;           // hid2 index
#pragma unroll
          for (int r = 0; r < 8; ++r) {
            int m = mt * 16 + r + hi8;              // edge row in chunk
            float v = acc[mt][nn][r] + sB2[n];
            v = v > 0.f ? v : 0.f;
            sFB[(n >> 4) * 512 + inv_b(m, n & 15)] = (_Float16)v;
          }
        }
    }
    __syncthreads();

    // Phase D: aggregate = escA(32x32) @ msg(32x256), accumulated in registers
    {
      v16h a0 = *(const v16h*)(sEA + 0 * 512 + lane * 16);
      v16h a1 = *(const v16h*)(sEA + 1 * 512 + lane * 16);
      v16h b0 = *(const v16h*)(sFB + (nt0)     * 512 + lane * 16);
      v16h b1 = *(const v16h*)(sFB + (nt0 + 1) * 512 + lane * 16);
      aggAcc[0][0] = WMMA_F16(a0, b0, aggAcc[0][0]);
      aggAcc[0][1] = WMMA_F16(a0, b1, aggAcc[0][1]);
      aggAcc[1][0] = WMMA_F16(a1, b0, aggAcc[1][0]);
      aggAcc[1][1] = WMMA_F16(a1, b1, aggAcc[1][1]);
    }
    __syncthreads();
  }

  // ---- output head ----
  // agg (registers) -> A-frags for out1
#pragma unroll
  for (int mt = 0; mt < 2; ++mt)
#pragma unroll
    for (int nn = 0; nn < 2; ++nn) {
      int k = (nt0 + nn) * 16 + ncol;               // hidden index = out1 K
#pragma unroll
      for (int r = 0; r < 8; ++r) {
        int m_loc = r + hi8;                        // node row within tile
        sFA[(mt * 8 + (k >> 5)) * 512 + inv_a(m_loc, k & 31)] =
            (_Float16)aggAcc[mt][nn][r];
      }
    }
  __syncthreads();

  // out1: aug @ W_out1 (agg part on WMMA, 5-lag part folded into epilogue)
  {
    v8f acc[2][2] = {{z8, z8}, {z8, z8}};
    wave_gemm(sFA, pO1, nt0, lane, acc);
#pragma unroll
    for (int mt = 0; mt < 2; ++mt)
#pragma unroll
      for (int nn = 0; nn < 2; ++nn) {
        int n = (nt0 + nn) * 16 + ncol;
#pragma unroll
        for (int r = 0; r < 8; ++r) {
          int m_loc = r + hi8;
          int m = mt * 16 + m_loc;                  // node
          float v = acc[mt][nn][r] + sBo1[n];
#pragma unroll
          for (int l = 0; l < LAG; ++l) v += sData[m * 8 + l] * sWo1d[l * HID + n];
          v = v > 0.f ? v : 0.f;
          sFB[(mt * 8 + (n >> 5)) * 512 + inv_a(m_loc, n & 31)] = (_Float16)v;
        }
      }
  }
  __syncthreads();

  // out2 GEMM + fused out3 dot-product epilogue
  {
    v8f acc[2][2] = {{z8, z8}, {z8, z8}};
    wave_gemm(sFB, pO2, nt0, lane, acc);
#pragma unroll
    for (int mt = 0; mt < 2; ++mt)
#pragma unroll
      for (int r = 0; r < 8; ++r) {
        int m = mt * 16 + r + hi8;                  // node
        float p = 0.f;
#pragma unroll
        for (int nn = 0; nn < 2; ++nn) {
          int n = (nt0 + nn) * 16 + ncol;
          float v = acc[mt][nn][r] + sBo2[n];
          v = v > 0.f ? v : 0.f;
          p += v * sWo3[n];
        }
        if (m < NA) atomicAdd(&sRed[m], p);
      }
  }
  __syncthreads();
  if (tid < NA) out[(b * NA + tid) * TMM + t] = sRed[tid] + b_out3[0];
}

// ---- host launcher --------------------------------------------------------
extern "C" void kernel_launch(void* const* d_in, const int* in_sizes, int n_in,
                              void* d_out, int out_size, void* d_ws, size_t ws_size,
                              hipStream_t stream) {
  (void)in_sizes; (void)n_in; (void)out_size; (void)ws_size;
  const float* inputs = (const float*)d_in[0];
  const float* mu     = (const float*)d_in[1];
  const float* logvar = (const float*)d_in[2];
  const float* eps    = (const float*)d_in[3];
  const float* w_msg1 = (const float*)d_in[6];
  const float* b_msg1 = (const float*)d_in[7];
  const float* w_msg2 = (const float*)d_in[8];
  const float* b_msg2 = (const float*)d_in[9];
  const float* w_out1 = (const float*)d_in[10];
  const float* b_out1 = (const float*)d_in[11];
  const float* w_out2 = (const float*)d_in[12];
  const float* b_out2 = (const float*)d_in[13];
  const float* w_out3 = (const float*)d_in[14];
  const float* b_out3 = (const float*)d_in[15];
  float* out = (float*)d_out;

  char* ws = (char*)d_ws;
  float*    esc = (float*)ws;                          // 6960 B (pad to 8192)
  _Float16* pW2 = (_Float16*)(ws + 8192);              // 128 KB
  _Float16* pO1 = (_Float16*)(ws + 8192 + 131072);     // 128 KB
  _Float16* pO2 = (_Float16*)(ws + 8192 + 262144);     // 128 KB
  _Float16* pW1 = (_Float16*)(ws + 8192 + 393216);     // 16 KB

  edge_scale_kernel<<<(BB * NE + 255) / 256, 256, 0, stream>>>(mu, logvar, eps, esc);
  pack_weights_kernel<<<400, 32, 0, stream>>>(w_msg1, w_msg2, w_out1, w_out2,
                                              pW1, pW2, pO1, pO2);

  dim3 grid(TMM, BB);   // 195 x 2 = 390 workgroups, one per (t, b)
  nri_decoder_kernel<<<grid, 256, SMEM_BYTES, stream>>>(
      inputs, esc, b_msg1, b_msg2,
      w_out1, b_out1, b_out2, w_out3, b_out3,
      pW1, pW2, pO1, pO2, out);
}
